// K_DAE_78142634983425
// MI455X (gfx1250) — matrix operand order repaired and checked by reading, not compiled, not run on hardware
//
#include <hip/hip_runtime.h>
#include <hip/hip_bf16.h>

// ---------------------------------------------------------------------------
// Problem constants
// ---------------------------------------------------------------------------
#define BATCH 32768
#define KC    10
#define D_    784
#define H1_   256
#define H2_   64
#define L_    16
#define KP0   800   // D padded to multiple of 32
#define KPD0  32    // L padded to multiple of 32

// bf16 transposed-weight blob layout (element offsets, per cluster)
#define OFF_E0 0u            // [256][800]
#define OFF_E1 204800u       // [ 64][256]
#define OFF_E2 221184u       // [ 16][ 64]
#define OFF_D0 222208u       // [ 64][ 32]
#define OFF_D1 224256u       // [256][ 64]
#define OFF_D2 240640u       // [784][256]
#define CL_STRIDE 441344u    // total bf16 elems per cluster

typedef __attribute__((ext_vector_type(16))) __bf16 bf16x16;
typedef __attribute__((ext_vector_type(8)))  float  f32x8;

union AFrag {
    bf16x16 v;
    uint4   q[2];
    __bf16  e[16];
};

// ---------------------------------------------------------------------------
// Fragment helpers (wave32 WMMA layouts per CDNA5 ISA 7.12.2)
// A 16x32 bf16: lane<16 -> M=lane, elems {0..7}=K+0..7, {8..15}=K+16..23
//               lane>=16 -> M=lane-16, elems {0..7}=K+8..15, {8..15}=K+24..31
// B 32x16 bf16: N=lane%16, elems in the same K order as A
// C 16x16 f32 : elem r -> M = r + 8*(lane>=16), N = lane%16
// ---------------------------------------------------------------------------
__device__ inline bf16x16 ldsA(const __bf16* a, int stride, int lane, int kbase) {
    int lr = lane & 15;
    int khalf = (lane >> 4) * 8;
    AFrag f;
    const __bf16* p = a + lr * stride + kbase + khalf;
    f.q[0] = *(const uint4*)(p);
    f.q[1] = *(const uint4*)(p + 16);
    return f.v;
}

__device__ inline bf16x16 gblB(const __bf16* wT, int Kp, int lane, int kbase) {
    int lc = lane & 15;
    int khalf = (lane >> 4) * 8;
    AFrag f;
    const uint4* p = (const uint4*)(wT + (size_t)lc * Kp + kbase + khalf);
    f.q[0] = p[0];
    f.q[1] = p[2];
    return f.v;
}

template <bool RELU>
__device__ inline void storeC(f32x8 c, __bf16* outTile, int stride, int lane,
                              const float* bias, int nbase) {
    int col = nbase + (lane & 15);
    float bv = bias[col];
    int rbase = (lane >> 4) * 8;
#pragma unroll
    for (int r = 0; r < 8; ++r) {
        float v = c[r] + bv;
        if (RELU) v = fmaxf(v, 0.0f);
        outTile[(rbase + r) * stride + col] = (__bf16)v;
    }
}

// Generic LDS-fed layer.  A fragments for the wave's whole 16xK tile are
// hoisted into registers once (<= 64 VGPRs for K<=256); each N-tile then
// batch-loads all its B fragments before the WMMA chain so the loads issue
// as one clause and the WMMAs run back-to-back on the matrix pipe.
template <int KSTEPS, int NTILES, bool RELU>
__device__ inline void layerLds(const __bf16* aTile, int aStride,
                                const __bf16* wT, int Kp, const float* bias,
                                __bf16* outTile, int outStride,
                                int ns, int lane) {
    bf16x16 aF[KSTEPS];
#pragma unroll
    for (int ks = 0; ks < KSTEPS; ++ks)
        aF[ks] = ldsA(aTile, aStride, lane, ks * 32);
    for (int nt = ns; nt < NTILES; nt += 2) {
        bf16x16 bF[KSTEPS];
#pragma unroll
        for (int ks = 0; ks < KSTEPS; ++ks)
            bF[ks] = gblB(wT + (size_t)(nt * 16) * Kp, Kp, lane, ks * 32);
        f32x8 acc = {};
#pragma unroll
        for (int ks = 0; ks < KSTEPS; ++ks)
            acc = __builtin_amdgcn_wmma_f32_16x16x32_bf16(
                false, aF[ks], false, bF[ks], (short)0, acc, false, false);
        storeC<RELU>(acc, outTile, outStride, lane, bias, nt * 16);
    }
}

// ---------------------------------------------------------------------------
// Kernel 1: per-cluster row index lists
// ---------------------------------------------------------------------------
__global__ void K_DAE_build_idx(const int* __restrict__ label,
                                int* __restrict__ counts,
                                int* __restrict__ idxList) {
    int b = blockIdx.x * blockDim.x + threadIdx.x;
    if (b >= BATCH) return;
    int c = label[b];
    int pos = atomicAdd(&counts[c], 1);
    idxList[c * BATCH + pos] = b;
}

// ---------------------------------------------------------------------------
// Kernel 2: fp32 [K][Kdim][N] -> bf16 transposed [K][N][Kp] (zero padded K)
// ---------------------------------------------------------------------------
__global__ void K_DAE_convert_wT(const float* __restrict__ w,
                                 __bf16* __restrict__ dst,
                                 int Kdim, int N, int Kp, unsigned layerOff) {
    size_t tid = (size_t)blockIdx.x * blockDim.x + threadIdx.x;
    size_t per = (size_t)N * Kp;
    size_t total = per * KC;
    if (tid >= total) return;
    int c = (int)(tid / per);
    size_t r = tid % per;
    int n = (int)(r / Kp);
    int k = (int)(r % Kp);
    float v = (k < Kdim) ? w[(size_t)c * Kdim * N + (size_t)k * N + n] : 0.0f;
    dst[(size_t)c * CL_STRIDE + layerOff + (size_t)n * Kp + k] = (__bf16)v;
}

// ---------------------------------------------------------------------------
// Kernel 3: fused 6-layer autoencoder, 64 gathered rows per workgroup
// ---------------------------------------------------------------------------
__global__ __launch_bounds__(256)
void K_DAE_fused(const float* __restrict__ x,
                 const float* __restrict__ be0, const float* __restrict__ be1,
                 const float* __restrict__ be2, const float* __restrict__ bd0,
                 const float* __restrict__ bd1, const float* __restrict__ bd2,
                 const int* __restrict__ counts, const int* __restrict__ idxList,
                 const __bf16* __restrict__ wbf, float* __restrict__ out) {
    __shared__ __bf16 sH1[64 * H1_];   // 32 KB (e0 out / d1 out)
    __shared__ __bf16 sH2[64 * H2_];   //  8 KB (e1 out / d0 out)
    __shared__ __bf16 sZ [64 * KPD0];  //  4 KB (e2 out, K-padded)
    __shared__ int    sCounts[KC];

    int tid = threadIdx.x;
    if (tid < KC) sCounts[tid] = counts[tid];
    for (int i = tid; i < 64 * KPD0; i += 256) sZ[i] = (__bf16)0.0f;
    __syncthreads();

    // resolve (cluster, local tile) for this block — uniform across block
    int t = blockIdx.x;
    int myC = -1, localT = 0, tb = 0;
    for (int c = 0; c < KC; ++c) {
        int ntc = (sCounts[c] + 63) >> 6;
        if (myC < 0 && t < tb + ntc) { myC = c; localT = t - tb; }
        tb += ntc;
    }
    if (myC < 0) return;
    int count = sCounts[myC];
    int rowBase = localT * 64;

    int wave = tid >> 5, lane = tid & 31;
    int mt = wave >> 1;    // M-tile 0..3 (16 rows each)
    int ns = wave & 1;     // N-tile split
    int lr = lane & 15;
    int hi = lane >> 4;

    const int* myIdx = idxList + myC * BATCH;

    // gathered row for A loads (clamped duplicate for padding rows)
    int pLoad = rowBase + mt * 16 + lr;
    int gLoad = myIdx[pLoad < count ? pLoad : (count - 1)];

    // gathered rows for the final scatter store
    int  gStore[8];
    bool vStore[8];
#pragma unroll
    for (int r = 0; r < 8; ++r) {
        int p = rowBase + mt * 16 + hi * 8 + r;
        vStore[r] = (p < count);
        gStore[r] = myIdx[p < count ? p : (count - 1)];
    }

    const __bf16* wc = wbf + (size_t)myC * CL_STRIDE;
    const float* be0c = be0 + myC * H1_;
    const float* be1c = be1 + myC * H2_;
    const float* be2c = be2 + myC * L_;
    const float* bd0c = bd0 + myC * H2_;
    const float* bd1c = bd1 + myC * H1_;
    const float* bd2c = bd2 + myC * D_;

    // ---- e0: x[gather] (784 -> 256), A from global fp32 with cvt ----
    {
        f32x8 acc[8] = {};
        const float* xr = x + (size_t)gLoad * D_;
        const __bf16* wT = wc + OFF_E0;
        for (int ks = 0; ks < 25; ++ks) {
            int kbase = ks * 32;
            AFrag a;
            const float* pA = xr + kbase + hi * 8;
            float4 f0 = *(const float4*)(pA);
            float4 f1 = *(const float4*)(pA + 4);
            const float* ff0 = (const float*)&f0;
            const float* ff1 = (const float*)&f1;
#pragma unroll
            for (int i = 0; i < 4; ++i) {
                a.e[i]     = (__bf16)ff0[i];
                a.e[4 + i] = (__bf16)ff1[i];
            }
            if (kbase < 768) {   // K tail: elems 8..15 map past 783 at kbase=768
                float4 g0 = *(const float4*)(pA + 16);
                float4 g1 = *(const float4*)(pA + 20);
                const float* gg0 = (const float*)&g0;
                const float* gg1 = (const float*)&g1;
#pragma unroll
                for (int i = 0; i < 4; ++i) {
                    a.e[8 + i]  = (__bf16)gg0[i];
                    a.e[12 + i] = (__bf16)gg1[i];
                }
            } else {
#pragma unroll
                for (int i = 0; i < 8; ++i) a.e[8 + i] = (__bf16)0.0f;
            }
            // batch-load all 8 B fragments, then run the 8-WMMA chain
            bf16x16 bF[8];
#pragma unroll
            for (int j = 0; j < 8; ++j) {
                int nt = ns + 2 * j;
                bF[j] = gblB(wT + (size_t)(nt * 16) * KP0, KP0, lane, kbase);
            }
#pragma unroll
            for (int j = 0; j < 8; ++j) {
                acc[j] = __builtin_amdgcn_wmma_f32_16x16x32_bf16(
                    false, a.v, false, bF[j], (short)0, acc[j], false, false);
            }
        }
        __bf16* outT = sH1 + mt * 16 * H1_;
#pragma unroll
        for (int j = 0; j < 8; ++j)
            storeC<true>(acc[j], outT, H1_, lane, be0c, (ns + 2 * j) * 16);
    }
    __syncthreads();

    // ---- e1: 256 -> 64 ----
    layerLds<8, 4, true>(sH1 + mt * 16 * H1_, H1_, wc + OFF_E1, 256, be1c,
                         sH2 + mt * 16 * H2_, H2_, ns, lane);
    __syncthreads();

    // ---- e2: 64 -> 16 ----
    layerLds<2, 1, true>(sH2 + mt * 16 * H2_, H2_, wc + OFF_E2, 64, be2c,
                         sZ + mt * 16 * KPD0, KPD0, ns, lane);
    __syncthreads();

    // ---- d0: 16 -> 64 (K padded to 32 with zeros on both sides) ----
    layerLds<1, 4, true>(sZ + mt * 16 * KPD0, KPD0, wc + OFF_D0, KPD0, bd0c,
                         sH2 + mt * 16 * H2_, H2_, ns, lane);
    __syncthreads();

    // ---- d1: 64 -> 256 ----
    layerLds<2, 16, true>(sH2 + mt * 16 * H2_, H2_, wc + OFF_D1, 64, bd1c,
                          sH1 + mt * 16 * H1_, H1_, ns, lane);
    __syncthreads();

    // ---- d2: 256 -> 784, linear, fp32 scatter to out ----
    {
        const __bf16* wT = wc + OFF_D2;
        const __bf16* aT = sH1 + mt * 16 * H1_;
        // hoist the wave's whole 16x256 A tile into registers (64 VGPRs)
        bf16x16 aF[8];
#pragma unroll
        for (int ks = 0; ks < 8; ++ks)
            aF[ks] = ldsA(aT, H1_, lane, ks * 32);
        for (int nt = ns; nt < 49; nt += 2) {
            bf16x16 bF[8];
#pragma unroll
            for (int ks = 0; ks < 8; ++ks)
                bF[ks] = gblB(wT + (size_t)(nt * 16) * 256, 256, lane, ks * 32);
            f32x8 acc = {};
#pragma unroll
            for (int ks = 0; ks < 8; ++ks)
                acc = __builtin_amdgcn_wmma_f32_16x16x32_bf16(
                    false, aF[ks], false, bF[ks], (short)0, acc, false, false);
            int col = nt * 16 + lr;
            float bv = bd2c[col];
#pragma unroll
            for (int r = 0; r < 8; ++r) {
                if (vStore[r])
                    out[(size_t)gStore[r] * D_ + col] = acc[r] + bv;
            }
        }
    }
}

// ---------------------------------------------------------------------------
// Launcher
// ---------------------------------------------------------------------------
extern "C" void kernel_launch(void* const* d_in, const int* in_sizes, int n_in,
                              void* d_out, int out_size, void* d_ws, size_t ws_size,
                              hipStream_t stream) {
    const float* x     = (const float*)d_in[0];
    const int*   label = (const int*)  d_in[1];
    const float* w_e0  = (const float*)d_in[2];
    const float* b_e0  = (const float*)d_in[3];
    const float* w_e1  = (const float*)d_in[4];
    const float* b_e1  = (const float*)d_in[5];
    const float* w_e2  = (const float*)d_in[6];
    const float* b_e2  = (const float*)d_in[7];
    const float* w_d0  = (const float*)d_in[8];
    const float* b_d0  = (const float*)d_in[9];
    const float* w_d1  = (const float*)d_in[10];
    const float* b_d1  = (const float*)d_in[11];
    const float* w_d2  = (const float*)d_in[12];
    const float* b_d2  = (const float*)d_in[13];
    float* out = (float*)d_out;

    // workspace layout: [counts 64 ints][idx K*B ints][bf16 weight blob]
    int* counts  = (int*)d_ws;
    int* idxList = counts + 64;
    __bf16* wbf  = (__bf16*)(idxList + (size_t)KC * BATCH);

    hipMemsetAsync(counts, 0, 64 * sizeof(int), stream);
    K_DAE_build_idx<<<BATCH / 256, 256, 0, stream>>>(label, counts, idxList);

    struct { const float* w; unsigned off; int Kdim, N, Kp; } layers[6] = {
        { w_e0, OFF_E0, D_,  H1_, KP0  },
        { w_e1, OFF_E1, H1_, H2_, H1_  },
        { w_e2, OFF_E2, H2_, L_,  H2_  },
        { w_d0, OFF_D0, L_,  H2_, KPD0 },
        { w_d1, OFF_D1, H2_, H1_, H2_  },
        { w_d2, OFF_D2, H1_, D_,  H1_  },
    };
    for (int i = 0; i < 6; ++i) {
        size_t total = (size_t)KC * layers[i].N * layers[i].Kp;
        int blocks = (int)((total + 255) / 256);
        K_DAE_convert_wT<<<blocks, 256, 0, stream>>>(
            layers[i].w, wbf, layers[i].Kdim, layers[i].N, layers[i].Kp,
            layers[i].off);
    }

    // ceil(B/64) tiles + up to K remainder tiles
    int maxTiles = BATCH / 64 + KC;
    K_DAE_fused<<<maxTiles, 256, 0, stream>>>(
        x, b_e0, b_e1, b_e2, b_d0, b_d1, b_d2, counts, idxList, wbf, out);
}
